// AffinityLoss_79482664780125
// MI455X (gfx1250) — compile-verified, matrix operand order checked by reading, not run on hardware
//
#include <hip/hip_runtime.h>
#include <math.h>

// ---------------------------------------------------------------------------
// CDNA5 types for WMMA / TDM
// ---------------------------------------------------------------------------
typedef __bf16 bf16_t;
typedef __bf16 v16bf __attribute__((ext_vector_type(16)));
typedef float  v8f   __attribute__((ext_vector_type(8)));
typedef unsigned int u32x4 __attribute__((ext_vector_type(4)));
typedef unsigned int u32x8 __attribute__((ext_vector_type(8)));

__device__ __forceinline__ unsigned short f32_to_bf16_rne(float x) {
    unsigned u = __float_as_uint(x);
    unsigned r = ((u >> 16) & 1u) + 0x7FFFu;   // round to nearest even
    return (unsigned short)((u + r) >> 16);
}
__device__ __forceinline__ float bf16u_to_f32(unsigned short s) {
    return __uint_as_float(((unsigned)s) << 16);
}
__device__ __forceinline__ bf16_t u2bf(unsigned short s) {
    return __builtin_bit_cast(bf16_t, s);
}

// clamped half-pixel linear coords (matches jax.image.resize 'bilinear' upsample)
__device__ __forceinline__ void lincoord(float sf, int nmax, int& i0, int& i1, float& t) {
    float f = floorf(sf);
    t = sf - f;
    int i = (int)f;
    int a = i;     if (a < 0) a = 0; if (a > nmax - 1) a = nmax - 1;
    int b = i + 1; if (b < 0) b = 0; if (b > nmax - 1) b = nmax - 1;
    i0 = a; i1 = b;
}
__device__ __forceinline__ float bil(const float* __restrict__ p, int stride,
                                     int y0, int y1, int x0, int x1, float ty, float tx) {
    float a = p[y0 * stride + x0], b = p[y0 * stride + x1];
    float c = p[y1 * stride + x0], d = p[y1 * stride + x1];
    float top = a + (b - a) * tx;
    float bot = c + (d - c) * tx;
    return top + (bot - top) * ty;
}

// ---------------------------------------------------------------------------
// K1: batch prototype sums over encoder (4,256,16,16), labels nearest@16
// grid = K*Ce = 1024 blocks, 256 threads
// ---------------------------------------------------------------------------
__global__ __launch_bounds__(256) void k_bp(const float* __restrict__ enc,
                                            const int* __restrict__ sup,
                                            float* __restrict__ bpS, float* __restrict__ bpC) {
    __shared__ float rs[256], rc[256];
    const int b = blockIdx.x, k = b >> 8, c = b & 255, t = threadIdx.x;
    float s = 0.f, cnt = 0.f;
    for (int i = t; i < 1024; i += 256) {              // n*256 + y*16 + x
        int n = i >> 8, rem = i & 255, y = rem >> 4, x = rem & 15;
        int lbl = sup[n * 65536 + y * 4096 + x * 16];  // sup[n, y*16, x*16]
        if (lbl == k) { s += enc[n * 65536 + c * 256 + y * 16 + x]; cnt += 1.f; }
    }
    rs[t] = s; rc[t] = cnt; __syncthreads();
    for (int st = 128; st > 0; st >>= 1) {
        if (t < st) { rs[t] += rs[t + st]; rc[t] += rc[t + st]; }
        __syncthreads();
    }
    if (t == 0) { bpS[k * 256 + c] = rs[0]; if (c == 0) bpC[k] = rc[0]; }
}

// ---------------------------------------------------------------------------
// K2: per-sample prototype sums over decoder (4,64,64,64), labels nearest@64
// grid = K*N*Cd = 1024 blocks, 256 threads
// ---------------------------------------------------------------------------
__global__ __launch_bounds__(256) void k_pp(const float* __restrict__ dec,
                                            const int* __restrict__ sup,
                                            float* __restrict__ ppS, float* __restrict__ ppC) {
    __shared__ float rs[256], rc[256];
    const int b = blockIdx.x, k = b >> 8, n = (b >> 6) & 3, c = b & 63, t = threadIdx.x;
    float s = 0.f, cnt = 0.f;
    for (int i = t; i < 4096; i += 256) {              // y*64 + x
        int y = i >> 6, x = i & 63;
        int lbl = sup[n * 65536 + y * 1024 + x * 4];   // sup[n, y*4, x*4]
        if (lbl == k) { s += dec[n * 262144 + c * 4096 + i]; cnt += 1.f; }
    }
    rs[t] = s; rc[t] = cnt; __syncthreads();
    for (int st = 128; st > 0; st >>= 1) {
        if (t < st) { rs[t] += rs[t + st]; rc[t] += rc[t + st]; }
        __syncthreads();
    }
    if (t == 0) { ppS[(k * 4 + n) * 64 + c] = rs[0]; if (c == 0) ppC[k * 4 + n] = rc[0]; }
}

// ---------------------------------------------------------------------------
// K3: assemble proto (K,N,320), L2-normalize, split to bf16 hi/lo.
// Packed per-sample slab for TDM: pHL u16 layout [n][{hi,lo}][kcol 0..15][c];
// kcol>=4 zero-padded.  grid = N*16 = 64 blocks, 128 threads
// ---------------------------------------------------------------------------
__global__ __launch_bounds__(128) void k_proto(const float* __restrict__ bpS, const float* __restrict__ bpC,
                                               const float* __restrict__ ppS, const float* __restrict__ ppC,
                                               unsigned short* __restrict__ pHL) {
    const int b = blockIdx.x, n = b >> 4, k = b & 15, t = threadIdx.x;
    __shared__ float red[128];
    __shared__ float pv[320];
    float ss = 0.f;
    if (k < 4) {
        for (int c = t; c < 320; c += 128) {
            float v;
            if (c < 256) v = bpS[k * 256 + c] / fmaxf(bpC[k], 1.0f);
            else         v = ppS[(k * 4 + n) * 64 + (c - 256)] / fmaxf(ppC[k * 4 + n], 1.0f);
            pv[c] = v; ss += v * v;
        }
    }
    red[t] = ss; __syncthreads();
    for (int st = 64; st > 0; st >>= 1) {
        if (t < st) red[t] += red[t + st];
        __syncthreads();
    }
    float inv = 1.0f / fmaxf(sqrtf(red[0]), 1e-12f);
    unsigned short* hB = pHL + n * 10240 + k * 320;          // hi plane
    unsigned short* lB = pHL + n * 10240 + 5120 + k * 320;   // lo plane
    for (int c = t; c < 320; c += 128) {
        if (k < 4) {
            float v = pv[c] * inv;
            unsigned short hu = f32_to_bf16_rne(v);
            hB[c] = hu;
            lB[c] = f32_to_bf16_rne(v - bf16u_to_f32(hu));
        } else { hB[c] = 0; lB[c] = 0; }
    }
}

// ---------------------------------------------------------------------------
// K4: rgb features = 2x2 avg-pool(sample)/0.1 -> (4,3,128,128)
// ---------------------------------------------------------------------------
__global__ __launch_bounds__(256) void k_pool(const float* __restrict__ sample,
                                              float* __restrict__ rgbf) {
    int i = blockIdx.x * 256 + threadIdx.x;
    if (i >= 4 * 3 * 128 * 128) return;
    int w = i & 127, h = (i >> 7) & 127, q = i >> 14, c = q % 3, n = q / 3;
    const float* p = sample + (n * 3 + c) * 65536;
    int y = 2 * h, x = 2 * w;
    float v = (p[y * 256 + x] + p[y * 256 + x + 1] +
               p[(y + 1) * 256 + x] + p[(y + 1) * 256 + x + 1]) * 0.25f;
    rgbf[i] = v * 10.0f;   // / SIGMA_RGB
}

// ---------------------------------------------------------------------------
// K5: fused bilinear-upsample + cosine projection (WMMA, bf16 hi/lo split)
//     + relu/affinity + y_redefined + class softmax.
// grid = N*H = 512 blocks; block = 256 threads = 8 waves; one wave per
// 16-pixel tile (block covers one image row).
// Prototype slab (20 KB/sample) is DMA'd into LDS once per block via the
// Tensor Data Mover (tensor_load_to_lds, TENSORcnt), then B fragments are
// fed from LDS. WMMA 16x16x32 bf16, error-compensated hi/lo split.
// ---------------------------------------------------------------------------
__global__ __launch_bounds__(256) void k_proj(const float* __restrict__ enc,
                                              const float* __restrict__ dec,
                                              const float* __restrict__ yhat,
                                              const unsigned short* __restrict__ pHL,
                                              float* __restrict__ yred,
                                              float* __restrict__ ysm) {
    __shared__ unsigned short sPro[10240];    // [hi|lo][16][320] u16 = 20 KB
    __shared__ float sdot[8][16][4];
    __shared__ float snrm[8][32];

    const int tid = threadIdx.x, wave = tid >> 5, lane = tid & 31;
    const int n = blockIdx.x >> 7, h = blockIdx.x & 127;
    const int w = (wave << 4) | (lane & 15);
    const int hi8 = (lane >= 16) ? 8 : 0;

    // --- TDM: DMA this sample's proto slab (20480 B) into LDS (wave 0 only) ---
    if (tid < 32) {
        unsigned long long ga = (unsigned long long)(const void*)(pHL + n * 10240);
        unsigned ldsOff = (unsigned)(unsigned long long)(void*)&sPro[0]; // addr[31:0] = LDS offset
        const unsigned td0 = 10240u;          // elements of 2 bytes
        u32x4 g0;
        g0[0] = 1u;                                           // count=1, user mode
        g0[1] = ldsOff;                                       // lds_addr
        g0[2] = (unsigned)(ga & 0xFFFFFFFFull);               // global_addr[31:0]
        g0[3] = ((unsigned)(ga >> 32) & 0x01FFFFFFu) | 0x80000000u; // addr[56:32] | type=2
        u32x8 g1;
        g1[0] = 0x00010000u;                                  // data_size=1 (2 bytes)
        g1[1] = (td0 & 0xFFFFu) << 16;                        // tensor_dim0 lo16
        g1[2] = (td0 >> 16) | (1u << 16);                     // tensor_dim0 hi | tensor_dim1=1 lo
        g1[3] = (td0 << 16);                                  // tensor_dim1 hi | tile_dim0
        g1[4] = 0u;                                           // tile_dim1=0, tile_dim2=0 (1-D tile)
        g1[5] = td0;                                          // tensor_dim0_stride lo32
        g1[6] = 0u;
        g1[7] = 0u;
        asm volatile("tensor_load_to_lds %0, %1" :: "s"(g0), "s"(g1) : "memory");
        __builtin_amdgcn_s_wait_tensorcnt(0);
    }
    __syncthreads();

    int ye0, ye1, xe0, xe1, yd0, yd1, xd0, xd1;
    float tye, txe, tyd, txd;
    lincoord((h + 0.5f) * 0.125f - 0.5f, 16, ye0, ye1, tye);
    lincoord((w + 0.5f) * 0.125f - 0.5f, 16, xe0, xe1, txe);
    lincoord((h + 0.5f) * 0.5f   - 0.5f, 64, yd0, yd1, tyd);
    lincoord((w + 0.5f) * 0.5f   - 0.5f, 64, xd0, xd1, txd);

    const float* encB = enc + n * 65536;
    const float* decB = dec + n * 262144;
    const int kb = (lane & 15) * 320;         // this lane's B column (class)

    v8f acc = {};
    float sumsq = 0.f;

    // encoder channels: chunks j=0..7 (c = 32j + off, off in lane's K-mapping)
    for (int j = 0; j < 8; ++j) {
        v16bf ahi = {}, alo = {}, bhi = {}, blo = {};
        #pragma unroll
        for (int e = 0; e < 16; ++e) {
            int off = (e & 7) + ((e >> 3) << 4) + hi8;     // ISA 16-bit A layout
            int c = 32 * j + off;
            float v = bil(encB + c * 256, 16, ye0, ye1, xe0, xe1, tye, txe);
            sumsq += v * v;
            unsigned short hu = f32_to_bf16_rne(v);
            ahi[e] = u2bf(hu);
            alo[e] = u2bf(f32_to_bf16_rne(v - bf16u_to_f32(hu)));
            bhi[e] = u2bf(sPro[kb + c]);
            blo[e] = u2bf(sPro[5120 + kb + c]);
        }
        acc = __builtin_amdgcn_wmma_f32_16x16x32_bf16(false, ahi, false, bhi, (short)0, acc, false, false);
        acc = __builtin_amdgcn_wmma_f32_16x16x32_bf16(false, ahi, false, blo, (short)0, acc, false, false);
        acc = __builtin_amdgcn_wmma_f32_16x16x32_bf16(false, alo, false, bhi, (short)0, acc, false, false);
    }
    // decoder channels: chunks j=0..1 (c = 256 + 32j + off)
    for (int j = 0; j < 2; ++j) {
        v16bf ahi = {}, alo = {}, bhi = {}, blo = {};
        #pragma unroll
        for (int e = 0; e < 16; ++e) {
            int off = (e & 7) + ((e >> 3) << 4) + hi8;
            int cd = 32 * j + off;
            int c = 256 + cd;
            float v = bil(decB + cd * 4096, 64, yd0, yd1, xd0, xd1, tyd, txd);
            sumsq += v * v;
            unsigned short hu = f32_to_bf16_rne(v);
            ahi[e] = u2bf(hu);
            alo[e] = u2bf(f32_to_bf16_rne(v - bf16u_to_f32(hu)));
            bhi[e] = u2bf(sPro[kb + c]);
            blo[e] = u2bf(sPro[5120 + kb + c]);
        }
        acc = __builtin_amdgcn_wmma_f32_16x16x32_bf16(false, ahi, false, bhi, (short)0, acc, false, false);
        acc = __builtin_amdgcn_wmma_f32_16x16x32_bf16(false, ahi, false, blo, (short)0, acc, false, false);
        acc = __builtin_amdgcn_wmma_f32_16x16x32_bf16(false, alo, false, bhi, (short)0, acc, false, false);
    }

    // lanes L and L+16 partition channels of pixel L%16 -> partial |f|^2 each
    snrm[wave][lane] = sumsq;
    if ((lane & 15) < 4) {                    // only classes 0..3 are real
        int k = lane & 15;
        int mb = (lane >= 16) ? 8 : 0;        // C/D layout: M = v + 8*[lane>=16]
        #pragma unroll
        for (int v = 0; v < 8; ++v) sdot[wave][mb + v][k] = acc[v];
    }
    __syncthreads();

    if (lane < 16) {                          // one lane per pixel of the tile
        int m = lane;
        float nrm = fmaxf(sqrtf(snrm[wave][m] + snrm[wave][m + 16]), 1e-12f);
        float cv[4], s = 0.f;
        #pragma unroll
        for (int k = 0; k < 4; ++k) { cv[k] = fmaxf(sdot[wave][m][k] / nrm, 0.f); s += cv[k]; }
        float invs = 1.0f / fmaxf(s, 1e-10f);
        int base = n * 65536 + h * 128 + w;
        float yr[4], mx = -1e30f;
        #pragma unroll
        for (int k = 0; k < 4; ++k) {
            float tdat = cv[k] * invs * yhat[base + k * 16384];
            yred[base + k * 16384] = tdat;    // output 2: y_redefined
            yr[k] = tdat; mx = fmaxf(mx, tdat);
        }
        float ev[4], es = 0.f;
        #pragma unroll
        for (int k = 0; k < 4; ++k) { ev[k] = expf(yr[k] - mx); es += ev[k]; }
        float ie = 1.0f / es;
        #pragma unroll
        for (int k = 0; k < 4; ++k) ysm[base + k * 16384] = ev[k] * ie;
    }
}

// ---------------------------------------------------------------------------
// K6: gated-CRF window term. One thread per pixel; 11x11 window, center
// skipped (reference zeroes it). Zero-padding semantics: OOB neighbor
// features are 0 (kernel = exp(-.5*|center|^2)), OOB y_sm contributes 0.
// Deterministic per-block partial sums for loss terms.
// grid = N*H = 512 blocks, 128 threads
// ---------------------------------------------------------------------------
__global__ __launch_bounds__(128) void k_crf(const float* __restrict__ rgbf,
                                             const float* __restrict__ ysm,
                                             const float* __restrict__ roi,
                                             float* __restrict__ prod_out,
                                             float* __restrict__ pK,
                                             float* __restrict__ pI,
                                             float* __restrict__ pR) {
    __shared__ float r0[128], r1[128], r2[128];
    const int n = blockIdx.x >> 7, h = blockIdx.x & 127, w = threadIdx.x;
    const float inv6 = 1.0f / 6.0f;
    const float fcx = w * inv6, fcy = h * inv6;
    const float* rg = rgbf + n * 3 * 16384;
    const int o0 = h * 128 + w;
    const float fr = rg[o0], fg = rg[16384 + o0], fb = rg[32768 + o0];
    const float* ys = ysm + n * 4 * 16384;
    const float y0 = ys[o0], y1 = ys[16384 + o0], y2 = ys[32768 + o0], y3 = ys[49152 + o0];

    float ksum = 0.f, pr0 = 0.f, pr1 = 0.f, pr2 = 0.f, pr3 = 0.f;
    for (int dy = -5; dy <= 5; ++dy) {
        for (int dx = -5; dx <= 5; ++dx) {
            if (dy == 0 && dx == 0) continue;
            int hn = h + dy, wn = w + dx;
            bool ib = ((unsigned)hn < 128u) && ((unsigned)wn < 128u);
            float d0, d1, d2, d3, d4;
            int o = hn * 128 + wn;
            if (ib) {
                d0 = dx * inv6; d1 = dy * inv6;
                d2 = rg[o] - fr; d3 = rg[16384 + o] - fg; d4 = rg[32768 + o] - fb;
            } else {
                d0 = -fcx; d1 = -fcy; d2 = -fr; d3 = -fg; d4 = -fb;
            }
            float kv = expf(-0.5f * (d0 * d0 + d1 * d1 + d2 * d2 + d3 * d3 + d4 * d4));
            ksum += kv;
            if (ib) {
                pr0 += kv * ys[o];
                pr1 += kv * ys[16384 + o];
                pr2 += kv * ys[32768 + o];
                pr3 += kv * ys[49152 + o];
            }
        }
    }
    const int pb = n * 65536 + o0;
    prod_out[pb]          = pr0;              // output 3: prod
    prod_out[pb + 16384]  = pr1;
    prod_out[pb + 32768]  = pr2;
    prod_out[pb + 49152]  = pr3;
    const float rv = roi[n * 16384 + o0];
    const float inner = rv * (pr0 * y0 + pr1 * y1 + pr2 * y2 + pr3 * y3);

    r0[w] = ksum; r1[w] = inner; r2[w] = rv; __syncthreads();
    for (int st = 64; st > 0; st >>= 1) {
        if (w < st) { r0[w] += r0[w + st]; r1[w] += r1[w + st]; r2[w] += r2[w + st]; }
        __syncthreads();
    }
    if (w == 0) { pK[blockIdx.x] = r0[0]; pI[blockIdx.x] = r1[0]; pR[blockIdx.x] = r2[0]; }
}

// ---------------------------------------------------------------------------
// K7: deterministic finalize: loss = max(sum(k) - sum(inner), 1e-5)/sum(roi)
// ---------------------------------------------------------------------------
__global__ __launch_bounds__(256) void k_final(const float* __restrict__ pK,
                                               const float* __restrict__ pI,
                                               const float* __restrict__ pR,
                                               float* __restrict__ out) {
    __shared__ float r0[256], r1[256], r2[256];
    int t = threadIdx.x;
    r0[t] = pK[t] + pK[t + 256];
    r1[t] = pI[t] + pI[t + 256];
    r2[t] = pR[t] + pR[t + 256];
    __syncthreads();
    for (int st = 128; st > 0; st >>= 1) {
        if (t < st) { r0[t] += r0[t + st]; r1[t] += r1[t + st]; r2[t] += r2[t + st]; }
        __syncthreads();
    }
    if (t == 0) out[0] = fmaxf(r0[0] - r1[0], 1e-5f) / r2[0];
}

// ---------------------------------------------------------------------------
extern "C" void kernel_launch(void* const* d_in, const int* in_sizes, int n_in,
                              void* d_out, int out_size, void* d_ws, size_t ws_size,
                              hipStream_t stream) {
    (void)in_sizes; (void)n_in; (void)out_size; (void)ws_size;
    const float* enc  = (const float*)d_in[0];   // (4,256,16,16)
    const float* dec  = (const float*)d_in[1];   // (4,64,64,64)
    const int*   sup  = (const int*)  d_in[2];   // (4,256,256)
    const float* yhat = (const float*)d_in[3];   // (4,4,128,128)
    const float* samp = (const float*)d_in[4];   // (4,3,256,256)
    const float* roi  = (const float*)d_in[5];   // (4,1,128,128)

    char* ws = (char*)d_ws;
    float*          bpS  = (float*)(ws + 0);                       // 4*256 f
    float*          bpC  = (float*)(ws + 4096);                    // 4 f
    float*          ppS  = (float*)(ws + 8192);                    // 4*4*64 f
    float*          ppC  = (float*)(ws + 12288);                   // 16 f
    unsigned short* pHL  = (unsigned short*)(ws + 16384);          // 4*2*16*320 u16 (81920 B)
    float*          ysm  = (float*)(ws + 98304);                   // 4*4*128*128 f
    float*          rgbf = (float*)(ws + 98304 + 1048576);         // 4*3*128*128 f
    float*          pK   = (float*)(ws + 98304 + 1048576 + 786432);// 512 f
    float*          pI   = pK + 512;
    float*          pR   = pK + 1024;

    float* out  = (float*)d_out;
    float* yred = out + 1;           // (4,4,128,128)
    float* prod = out + 1 + 262144;  // (4,4,128,128)

    k_bp   <<<dim3(1024), dim3(256), 0, stream>>>(enc, sup, bpS, bpC);
    k_pp   <<<dim3(1024), dim3(256), 0, stream>>>(dec, sup, ppS, ppC);
    k_proto<<<dim3(64),   dim3(128), 0, stream>>>(bpS, bpC, ppS, ppC, pHL);
    k_pool <<<dim3(768),  dim3(256), 0, stream>>>(samp, rgbf);
    k_proj <<<dim3(512),  dim3(256), 0, stream>>>(enc, dec, yhat, pHL, yred, ysm);
    k_crf  <<<dim3(512),  dim3(128), 0, stream>>>(rgbf, ysm, roi, prod, pK, pI, pR);
    k_final<<<dim3(1),    dim3(256), 0, stream>>>(pK, pI, pR, out);
}
